// TransformerBlock_76476187672712
// MI455X (gfx1250) — matrix operand order repaired and checked by reading, not compiled, not run on hardware
//
#include <hip/hip_runtime.h>
#include <hip/hip_bf16.h>
#include <math.h>

typedef __attribute__((ext_vector_type(16))) __bf16 v16bf;
typedef __attribute__((ext_vector_type(8)))  float  v8f;

#define DMODEL 768
#define NHEAD  12
#define HDIM   64
#define DFF    3072
#define DQKV   2304

// ---------- helpers ----------
static __device__ __forceinline__ unsigned short f2bf(float f) {
  union { float f; unsigned int u; } c; c.f = f;
  unsigned int u = c.u;
  unsigned int r = u + 0x7FFFu + ((u >> 16) & 1u);   // round-to-nearest-even
  return (unsigned short)(r >> 16);
}

union FragU { uint4 q[2]; v16bf v; };

static __device__ __forceinline__ v8f zero_v8f() {
  v8f z = {0.f,0.f,0.f,0.f,0.f,0.f,0.f,0.f};
  return z;
}

// A fragment: 16(M) x 32(K) bf16, source row-major [16][ld]
// lanes 0-15: e0..7 -> K=0..7, e8..15 -> K=16..23 ; lanes 16-31: +8 (ISA 7.12.2)
static __device__ __forceinline__ v16bf load_a_frag(const unsigned short* base, int ld) {
  int lane = threadIdx.x & 31;
  int m = lane & 15, half = lane >> 4;
  const unsigned short* p = base + (size_t)m * ld;
  FragU f;
  f.q[0] = *(const uint4*)(p + half * 8);
  f.q[1] = *(const uint4*)(p + 16 + half * 8);
  return f.v;
}

// B fragment: 32(K) x 16(N) bf16, source stored column-major [N][ld over K]
// lanes 0-15: e0..7 -> K=0..7, e8..15 -> K=16..23 ; lanes 16-31: +8 ; N=lane%16
static __device__ __forceinline__ v16bf load_b_frag(const unsigned short* base, int ld) {
  int lane = threadIdx.x & 31;
  int n = lane & 15, half = lane >> 4;
  const unsigned short* p = base + (size_t)n * ld;
  FragU f;
  f.q[0] = *(const uint4*)(p + half * 8);
  f.q[1] = *(const uint4*)(p + 16 + half * 8);
  return f.v;
}

static __device__ __forceinline__ v8f wmma_bf16(v16bf a, v16bf b, v8f c) {
  return __builtin_amdgcn_wmma_f32_16x16x32_bf16(false, a, false, b, (short)0, c, false, false);
}

// ---------- fp32 -> bf16 weight conversion WITH transpose ----------
// src [rows][cols] f32 ; dst stores B^T: dst[(dstRowOff + c) * rows + r]
// (rows == K, so GEMM can stage B with contiguous 128-bit loads along K)
__global__ __launch_bounds__(256) void cvt_f32_bf16_transpose(const float* __restrict__ src,
    unsigned short* __restrict__ dst, int rows, int cols, int dstRowOff) {
  int i = blockIdx.x * 256 + threadIdx.x;
  if (i >= rows * cols) return;
  int r = i / cols, c = i - r * cols;
  dst[(size_t)(dstRowOff + c) * rows + r] = f2bf(src[i]);
}

// ---------- LayerNorm (row per block) -> bf16 ----------
__global__ __launch_bounds__(256) void layernorm_to_bf16(const float* __restrict__ x,
    const float* __restrict__ g, const float* __restrict__ b,
    unsigned short* __restrict__ out) {
  int row = blockIdx.x;
  const float* xr = x + (size_t)row * DMODEL;
  __shared__ float red[8];
  __shared__ float stat[2];
  int tid = threadIdx.x;
  float s = 0.f;
  for (int i = tid; i < DMODEL; i += 256) s += xr[i];
  for (int o = 16; o > 0; o >>= 1) s += __shfl_xor(s, o, 32);
  if ((tid & 31) == 0) red[tid >> 5] = s;
  __syncthreads();
  if (tid == 0) { float t = 0.f; for (int i = 0; i < 8; ++i) t += red[i]; stat[0] = t * (1.f / DMODEL); }
  __syncthreads();
  float mean = stat[0];
  float v = 0.f;
  for (int i = tid; i < DMODEL; i += 256) { float d = xr[i] - mean; v += d * d; }
  for (int o = 16; o > 0; o >>= 1) v += __shfl_xor(v, o, 32);
  if ((tid & 31) == 0) red[tid >> 5] = v;
  __syncthreads();
  if (tid == 0) { float t = 0.f; for (int i = 0; i < 8; ++i) t += red[i]; stat[1] = rsqrtf(t * (1.f / DMODEL) + 1e-5f); }
  __syncthreads();
  float inv = stat[1];
  for (int i = tid; i < DMODEL; i += 256)
    out[(size_t)row * DMODEL + i] = f2bf(g[i] * (xr[i] - mean) * inv + b[i]);
}

// ---------- tiled bf16 WMMA GEMM: C = act(A@B + bias) (+resid) ----------
// A [M,K] row-major bf16 ; Bt = B^T [N,K] row-major bf16 (pre-transposed weights)
// 128x128 block tile, 8 waves (4x2), 32x64 per wave -> 8 WMMAs / k-step / wave.
// Register double-buffer: next k-tile global loads issue underneath the WMMAs.
// All 6 fragments (2 A + 4 B) are hoisted before the WMMA burst so the 12
// ds_load_b128 issue together and WMMAs drain with partial dscnt waits.
#define GBM 128
#define GBN 128
#define GBK 32
#define GLD 40   // padded K-stride in LDS: 80B row pitch (16B aligned), spreads banks

__global__ __launch_bounds__(256) void gemm_bf16_wmma(
    const unsigned short* __restrict__ A, const unsigned short* __restrict__ Bt,
    const float* __restrict__ bias, const float* __restrict__ resid,
    float* __restrict__ outF, unsigned short* __restrict__ outB,
    int M, int N, int K, int do_gelu) {
  __shared__ __align__(16) unsigned short sA[GBM * GLD];
  __shared__ __align__(16) unsigned short sB[GBN * GLD];
  int bm = blockIdx.y, bn = blockIdx.x;
  int tid = threadIdx.x, wave = tid >> 5, lane = tid & 31;
  int wm = wave >> 1, wn = wave & 1;
  int r0 = tid >> 2, c0 = (tid & 3) * 8;         // staging: rows r0 and r0+64, 8 bf16 at col c0
  (void)M;

  const unsigned short* Ag = A  + (size_t)(bm * GBM + r0) * K + c0;
  const unsigned short* Bg = Bt + (size_t)(bn * GBN + r0) * K + c0;

  v8f acc[2][4];
#pragma unroll
  for (int s = 0; s < 2; ++s)
#pragma unroll
    for (int t = 0; t < 4; ++t) acc[s][t] = zero_v8f();

  // prefetch first k-tile into registers
  uint4 ra0 = *(const uint4*)(Ag);
  uint4 ra1 = *(const uint4*)(Ag + (size_t)64 * K);
  uint4 rb0 = *(const uint4*)(Bg);
  uint4 rb1 = *(const uint4*)(Bg + (size_t)64 * K);

  for (int k0 = 0; k0 < K; k0 += GBK) {
    __syncthreads();                              // prior compute done reading LDS
    *(uint4*)&sA[r0 * GLD + c0]        = ra0;
    *(uint4*)&sA[(r0 + 64) * GLD + c0] = ra1;
    *(uint4*)&sB[r0 * GLD + c0]        = rb0;
    *(uint4*)&sB[(r0 + 64) * GLD + c0] = rb1;
    __syncthreads();

    int k1 = k0 + GBK;
    if (k1 < K) {                                 // issue next tile's global loads now
      ra0 = *(const uint4*)(Ag + k1);
      ra1 = *(const uint4*)(Ag + (size_t)64 * K + k1);
      rb0 = *(const uint4*)(Bg + k1);
      rb1 = *(const uint4*)(Bg + (size_t)64 * K + k1);
    }

    // hoist all fragment loads, then burst 8 WMMAs
    v16bf af0 = load_a_frag(&sA[(wm * 32) * GLD], GLD);
    v16bf af1 = load_a_frag(&sA[(wm * 32 + 16) * GLD], GLD);
    v16bf bf0 = load_b_frag(&sB[(wn * 64 +  0) * GLD], GLD);
    v16bf bf1 = load_b_frag(&sB[(wn * 64 + 16) * GLD], GLD);
    v16bf bf2 = load_b_frag(&sB[(wn * 64 + 32) * GLD], GLD);
    v16bf bf3 = load_b_frag(&sB[(wn * 64 + 48) * GLD], GLD);

    acc[0][0] = wmma_bf16(af0, bf0, acc[0][0]);
    acc[1][0] = wmma_bf16(af1, bf0, acc[1][0]);
    acc[0][1] = wmma_bf16(af0, bf1, acc[0][1]);
    acc[1][1] = wmma_bf16(af1, bf1, acc[1][1]);
    acc[0][2] = wmma_bf16(af0, bf2, acc[0][2]);
    acc[1][2] = wmma_bf16(af1, bf2, acc[1][2]);
    acc[0][3] = wmma_bf16(af0, bf3, acc[0][3]);
    acc[1][3] = wmma_bf16(af1, bf3, acc[1][3]);
  }

  // epilogue (C layout: VGPR r -> row = r + (lane/16)*8, col = lane%16)
  int half = lane >> 4, nl = lane & 15;
#pragma unroll
  for (int s = 0; s < 2; ++s) {
    int row0 = bm * GBM + wm * 32 + s * 16 + half * 8;
#pragma unroll
    for (int t = 0; t < 4; ++t) {
      int col = bn * GBN + wn * 64 + t * 16 + nl;
      float bb = bias ? bias[col] : 0.f;
#pragma unroll
      for (int r = 0; r < 8; ++r) {
        float vv = acc[s][t][r] + bb;
        if (do_gelu) vv = 0.5f * vv * (1.f + erff(vv * 0.70710678118f)); // exact GELU
        size_t idx = (size_t)(row0 + r) * N + col;
        if (resid) vv += resid[idx];
        if (outF) outF[idx] = vv;
        if (outB) outB[idx] = f2bf(vv);
      }
    }
  }
}

// ---------- causal flash attention, WMMA for QK^T and P@V ----------
#define ALDVT 40
__global__ __launch_bounds__(128) void attn_flash_wmma(
    const unsigned short* __restrict__ qkv, unsigned short* __restrict__ ctx, int T) {
  (void)T;
  int h = blockIdx.y;
  int qb = blockIdx.x;
  int tid = threadIdx.x, wave = tid >> 5, lane = tid & 31;
  int half = lane >> 4, nl = lane & 15;
  __shared__ __align__(16) unsigned short sK[32 * 64];      // [key][hd]
  __shared__ __align__(16) unsigned short sVt[64 * ALDVT];  // [hd][key] padded
  __shared__ __align__(16) float          sS[4][16 * 32];   // per-wave score scratch
  __shared__ __align__(16) unsigned short sP[4][16 * 32];   // per-wave bf16 probs
  __shared__ float sM[4][16], sL[4][16], sAl[4][16];

  int q0 = qb * 64 + wave * 16;
  const unsigned short* Qb = qkv + (size_t)q0 * DQKV + h * HDIM;
  v16bf qf0 = load_a_frag(Qb, DQKV);        // hd 0..31
  v16bf qf1 = load_a_frag(Qb + 32, DQKV);   // hd 32..63

  if (lane < 16) { sM[wave][lane] = -3.0e38f; sL[wave][lane] = 0.f; }
  v8f o[4] = { zero_v8f(), zero_v8f(), zero_v8f(), zero_v8f() };

  int kend = qb * 64 + 64;
  for (int kb = 0; kb < kend; kb += 32) {
    __syncthreads();
    // stage K [32][64] and V transposed [64][key]
    for (int i = tid; i < 32 * 8; i += 128) {
      int key = i >> 3, c = (i & 7) * 8;
      *(uint4*)&sK[key * 64 + c] =
          *(const uint4*)(qkv + (size_t)(kb + key) * DQKV + DMODEL + h * HDIM + c);
      uint4 d = *(const uint4*)(qkv + (size_t)(kb + key) * DQKV + 2 * DMODEL + h * HDIM + c);
      const unsigned short* e = (const unsigned short*)&d;
#pragma unroll
      for (int j = 0; j < 8; ++j) sVt[(c + j) * ALDVT + key] = e[j];
    }
    __syncthreads();

    // scores S = (Q K^T)/8 with causal mask -> per-wave LDS scratch
#pragma unroll
    for (int nt = 0; nt < 2; ++nt) {
      v16bf b0 = load_b_frag(&sK[nt * 16 * 64], 64);        // hd 0..31
      v16bf b1 = load_b_frag(&sK[nt * 16 * 64 + 32], 64);   // hd 32..63
      v8f s = zero_v8f();
      s = wmma_bf16(qf0, b0, s);
      s = wmma_bf16(qf1, b1, s);
#pragma unroll
      for (int r = 0; r < 8; ++r) {
        int qr = q0 + half * 8 + r;
        int kc = kb + nt * 16 + nl;
        float v = s[r] * 0.125f;                 // 1/sqrt(64)
        if (kc > qr) v = -3.0e38f;
        sS[wave][(half * 8 + r) * 32 + nt * 16 + nl] = v;
      }
    }

    // online softmax (wave-private LDS; same-wave DS ops are in-order)
    if (lane < 16) {
      int r = lane;
      float* Sr = &sS[wave][r * 32];
      unsigned short* Pr = &sP[wave][r * 32];
      float m_old = sM[wave][r];
      float m_new = m_old;
      for (int j = 0; j < 32; ++j) m_new = fmaxf(m_new, Sr[j]);
      float l = sL[wave][r];
      float alpha = 1.f;
      if (m_new < -1.0e37f) {                    // fully-masked block for this row
        for (int j = 0; j < 32; ++j) Pr[j] = 0;
      } else {
        alpha = __expf(m_old - m_new);
        l *= alpha;
        for (int j = 0; j < 32; ++j) { float pv = __expf(Sr[j] - m_new); l += pv; Pr[j] = f2bf(pv); }
      }
      sM[wave][r] = m_new; sL[wave][r] = l; sAl[wave][r] = alpha;
    }

    // rescale accumulator, then O += P @ V (all fragments hoisted)
#pragma unroll
    for (int t = 0; t < 4; ++t)
#pragma unroll
      for (int r = 0; r < 8; ++r) o[t][r] *= sAl[wave][half * 8 + r];

    v16bf pf  = load_a_frag(&sP[wave][0], 32);
    v16bf vf0 = load_b_frag(&sVt[ 0 * ALDVT], ALDVT);
    v16bf vf1 = load_b_frag(&sVt[16 * ALDVT], ALDVT);
    v16bf vf2 = load_b_frag(&sVt[32 * ALDVT], ALDVT);
    v16bf vf3 = load_b_frag(&sVt[48 * ALDVT], ALDVT);
    o[0] = wmma_bf16(pf, vf0, o[0]);
    o[1] = wmma_bf16(pf, vf1, o[1]);
    o[2] = wmma_bf16(pf, vf2, o[2]);
    o[3] = wmma_bf16(pf, vf3, o[3]);
  }

  // normalize and store ctx (bf16, [T, 768] interleaved heads)
  unsigned short* Cb = ctx + (size_t)q0 * DMODEL + h * HDIM;
#pragma unroll
  for (int t = 0; t < 4; ++t)
#pragma unroll
    for (int r = 0; r < 8; ++r) {
      float val = o[t][r] / sL[wave][half * 8 + r];
      Cb[(size_t)(half * 8 + r) * DMODEL + t * 16 + nl] = f2bf(val);
    }
}

// ---------- orchestration ----------
extern "C" void kernel_launch(void* const* d_in, const int* in_sizes, int n_in,
                              void* d_out, int out_size, void* d_ws, size_t ws_size,
                              hipStream_t stream) {
  (void)n_in; (void)out_size; (void)ws_size;
  const float* x   = (const float*)d_in[0];
  const float* Wq  = (const float*)d_in[1];
  const float* Wk  = (const float*)d_in[2];
  const float* Wv  = (const float*)d_in[3];
  const float* Wo  = (const float*)d_in[4];
  const float* bo  = (const float*)d_in[5];
  const float* W1  = (const float*)d_in[6];
  const float* b1  = (const float*)d_in[7];
  const float* W2  = (const float*)d_in[8];
  const float* b2  = (const float*)d_in[9];
  const float* g1  = (const float*)d_in[10];
  const float* be1 = (const float*)d_in[11];
  const float* g2  = (const float*)d_in[12];
  const float* be2 = (const float*)d_in[13];
  int T = in_sizes[0] / DMODEL;   // 4096

  char* p = (char*)d_ws;
  auto alloc = [&](size_t bytes) -> char* {
    char* r = p; p += (bytes + 255) & ~(size_t)255; return r;
  };
  // all weight buffers hold B^T [N][K] in bf16
  unsigned short* wqkv_t = (unsigned short*)alloc((size_t)DQKV * DMODEL * 2);   // [2304][768]
  unsigned short* wo_t   = (unsigned short*)alloc((size_t)DMODEL * DMODEL * 2); // [768][768]
  unsigned short* w1_t   = (unsigned short*)alloc((size_t)DFF * DMODEL * 2);    // [3072][768]
  unsigned short* w2_t   = (unsigned short*)alloc((size_t)DMODEL * DFF * 2);    // [768][3072]
  unsigned short* h1     = (unsigned short*)alloc((size_t)T * DMODEL * 2);      // reused as h2
  unsigned short* qkv    = (unsigned short*)alloc((size_t)T * DQKV * 2);
  unsigned short* ctxb   = (unsigned short*)alloc((size_t)T * DMODEL * 2);
  float*          x1     = (float*)alloc((size_t)T * DMODEL * 4);
  unsigned short* a1     = qkv;  // a1 [T,3072] overlays dead qkv[T,2304]+ctx[T,768] (contiguous)

  // weight conversions (+transpose), one-time
  {
    int n = DMODEL * DMODEL, g = (n + 255) / 256;
    cvt_f32_bf16_transpose<<<g, 256, 0, stream>>>(Wq, wqkv_t, DMODEL, DMODEL, 0);
    cvt_f32_bf16_transpose<<<g, 256, 0, stream>>>(Wk, wqkv_t, DMODEL, DMODEL, DMODEL);
    cvt_f32_bf16_transpose<<<g, 256, 0, stream>>>(Wv, wqkv_t, DMODEL, DMODEL, 2 * DMODEL);
    cvt_f32_bf16_transpose<<<g, 256, 0, stream>>>(Wo, wo_t, DMODEL, DMODEL, 0);
    int n2 = DMODEL * DFF, g2g = (n2 + 255) / 256;
    cvt_f32_bf16_transpose<<<g2g, 256, 0, stream>>>(W1, w1_t, DMODEL, DFF, 0);
    cvt_f32_bf16_transpose<<<g2g, 256, 0, stream>>>(W2, w2_t, DFF, DMODEL, 0);
  }

  // attention branch
  layernorm_to_bf16<<<T, 256, 0, stream>>>(x, g1, be1, h1);
  gemm_bf16_wmma<<<dim3(DQKV / GBN, T / GBM), 256, 0, stream>>>(
      h1, wqkv_t, nullptr, nullptr, nullptr, qkv, T, DQKV, DMODEL, 0);
  attn_flash_wmma<<<dim3(T / 64, NHEAD), 128, 0, stream>>>(qkv, ctxb, T);
  gemm_bf16_wmma<<<dim3(DMODEL / GBN, T / GBM), 256, 0, stream>>>(
      ctxb, wo_t, bo, x, x1, nullptr, T, DMODEL, DMODEL, 0);

  // FFN branch
  layernorm_to_bf16<<<T, 256, 0, stream>>>(x1, g2, be2, h1);
  gemm_bf16_wmma<<<dim3(DFF / GBN, T / GBM), 256, 0, stream>>>(
      h1, w1_t, b1, nullptr, nullptr, a1, T, DFF, DMODEL, 1);
  gemm_bf16_wmma<<<dim3(DMODEL / GBN, T / GBM), 256, 0, stream>>>(
      a1, w2_t, b2, x1, (float*)d_out, nullptr, T, DMODEL, DFF, 0);
}
